// InterpretableMultiHeadAttention_3959959847313
// MI455X (gfx1250) — matrix operand order repaired and checked by reading, not compiled, not run on hardware
//
#include <hip/hip_runtime.h>
#include <hip/hip_bf16.h>

typedef __attribute__((ext_vector_type(16))) __bf16 v16bf;
typedef __attribute__((ext_vector_type(8)))  __bf16 bf16x8;
typedef __attribute__((ext_vector_type(8)))  float  v8f;
typedef __attribute__((ext_vector_type(4)))  float  f32x4;
typedef __attribute__((ext_vector_type(4)))  unsigned u32x4;
typedef __attribute__((ext_vector_type(8)))  unsigned u32x8;

#define DEVFN __device__ __forceinline__

constexpr int BB  = 16;
constexpr int SS  = 1024;
constexpr int DD  = 256;
constexpr int HH  = 4;
constexpr int DKK = 64;
constexpr float NEG_INF = -1.0e9f;
constexpr float SCALE   = 0.125f;   // 1/sqrt(64)

// ---------------- WMMA fragment loaders (ISA 7.12.2 layouts, wave32) ----------------

DEVFN v16bf load_a_bf16(const __bf16* base, int row0, int ld, int kc, int lane) {
  const int lrow = lane & 15, half = lane >> 4;
  const __bf16* p = base + (size_t)(row0 + lrow) * ld + kc + half * 8;
  bf16x8 lo = *(const bf16x8*)p;
  bf16x8 hi = *(const bf16x8*)(p + 16);
  v16bf a;
#pragma unroll
  for (int i = 0; i < 8; ++i) { a[i] = lo[i]; a[8 + i] = hi[i]; }
  return a;
}

DEVFN v16bf load_a_f32(const float* base, int row0, int ld, int kc, int lane) {
  const int lrow = lane & 15, half = lane >> 4;
  const float* p = base + (size_t)(row0 + lrow) * ld + kc + half * 8;
  f32x4 c0 = *(const f32x4*)p;
  f32x4 c1 = *(const f32x4*)(p + 4);
  f32x4 c2 = *(const f32x4*)(p + 16);
  f32x4 c3 = *(const f32x4*)(p + 20);
  v16bf a;
#pragma unroll
  for (int i = 0; i < 4; ++i) {
    a[i]      = (__bf16)c0[i];  a[4 + i]  = (__bf16)c1[i];
    a[8 + i]  = (__bf16)c2[i];  a[12 + i] = (__bf16)c3[i];
  }
  return a;
}

// B operand stored transposed as [N][K] bf16 -> two 8-K chunks per lane (k=8h..+7, 16+8h..+7).
DEVFN v16bf load_b_bf16(const __bf16* baseT, int n0, int ld, int kc, int lane) {
  const int n = lane & 15, half = lane >> 4;
  const __bf16* p = baseT + (size_t)(n0 + n) * ld + kc + half * 8;
  bf16x8 lo = *(const bf16x8*)p;
  bf16x8 hi = *(const bf16x8*)(p + 16);
  v16bf b;
#pragma unroll
  for (int i = 0; i < 8; ++i) { b[i] = lo[i]; b[8 + i] = hi[i]; }
  return b;
}

DEVFN v8f wmma_bf16(v16bf a, v16bf b, v8f c) {
  return __builtin_amdgcn_wmma_f32_16x16x32_bf16(false, a, false, b, (short)0, c,
                                                 false, false);
}

DEVFN void wait_tensorcnt0() {
#if __has_builtin(__builtin_amdgcn_s_wait_tensorcnt)
  __builtin_amdgcn_s_wait_tensorcnt(0);
#else
  asm volatile("s_wait_tensorcnt 0x0" ::: "memory");
#endif
}

// ---------------- K0: weight transpose + bf16 convert ----------------
__global__ void prep_weights_kernel(const float* __restrict__ Wq,
                                    const float* __restrict__ Wk,
                                    const float* __restrict__ Wv,
                                    __bf16* __restrict__ WqT,
                                    __bf16* __restrict__ WkT,
                                    __bf16* __restrict__ WvT) {
  int i = blockIdx.x * blockDim.x + threadIdx.x;
  const int nqk = HH * DKK * DD;  // 65536
  if (i < nqk) {
    int h = i / (DKK * DD);
    int r = i % (DKK * DD);
    int n = r / DD, d = r % DD;
    size_t src = ((size_t)h * DD + d) * DKK + n;
    WqT[i] = (__bf16)Wq[src];
    WkT[i] = (__bf16)Wk[src];
  } else {
    int j = i - nqk;
    if (j < DKK * DD) {
      int n = j / DD, d = j % DD;
      WvT[j] = (__bf16)Wv[(size_t)d * DKK + n];
    }
  }
}

// ---------------- K1a: Q/K head projections via WMMA ----------------
__global__ void proj_qk_kernel(const float* __restrict__ q,
                               const float* __restrict__ k,
                               const __bf16* __restrict__ WqT,
                               const __bf16* __restrict__ WkT,
                               const float* __restrict__ bq,
                               const float* __restrict__ bk,
                               __bf16* __restrict__ qs,
                               __bf16* __restrict__ kkb) {
  const int tid = threadIdx.x, lane = tid & 31, wave = tid >> 5;
  const int h = blockIdx.y;
  const bool isK = blockIdx.z != 0;
  const float*  X    = isK ? k : q;
  const __bf16* WT   = (isK ? WkT : WqT) + (size_t)h * DKK * DD;
  const float*  bias = (isK ? bk : bq) + h * DKK;
  __bf16*       OUT  = isK ? kkb : qs;

  const int rowTile = blockIdx.x;
  const int b  = rowTile >> 6;
  const int s0 = (rowTile & 63) << 4;
  const int grow = rowTile << 4;

  v8f acc[2] = {v8f{}, v8f{}};
  for (int kc = 0; kc < DD; kc += 32) {
    v16bf a  = load_a_f32(X, grow, DD, kc, lane);
#pragma unroll
    for (int t = 0; t < 2; ++t) {
      v16bf bf = load_b_bf16(WT, (wave * 2 + t) * 16, DD, kc, lane);
      acc[t] = wmma_bf16(a, bf, acc[t]);
    }
  }
  const int n = lane & 15, half = lane >> 4;
  const size_t outBase = ((size_t)(b * HH + h) * SS + s0) * DKK;
#pragma unroll
  for (int t = 0; t < 2; ++t) {
    const int n0 = (wave * 2 + t) * 16;
    const float bb = bias[n0 + n];
#pragma unroll
    for (int r = 0; r < 8; ++r) {
      const int M = r + half * 8;
      OUT[outBase + (size_t)M * DKK + n0 + n] = (__bf16)(acc[t][r] + bb);
    }
  }
}

// ---------------- K1b: V projection via WMMA, stored transposed vsT[b][dk][S] ----------------
__global__ void proj_v_kernel(const float* __restrict__ v,
                              const __bf16* __restrict__ WvT,
                              const float* __restrict__ bv,
                              __bf16* __restrict__ vsT) {
  const int tid = threadIdx.x, lane = tid & 31, wave = tid >> 5;
  const int rowTile = blockIdx.x;
  const int b  = rowTile >> 6;
  const int s0 = (rowTile & 63) << 4;
  const int grow = rowTile << 4;

  v8f acc[2] = {v8f{}, v8f{}};
  for (int kc = 0; kc < DD; kc += 32) {
    v16bf a = load_a_f32(v, grow, DD, kc, lane);
#pragma unroll
    for (int t = 0; t < 2; ++t) {
      v16bf bf = load_b_bf16(WvT, (wave * 2 + t) * 16, DD, kc, lane);
      acc[t] = wmma_bf16(a, bf, acc[t]);
    }
  }
  const int n = lane & 15, half = lane >> 4;
#pragma unroll
  for (int t = 0; t < 2; ++t) {
    const int c = (wave * 2 + t) * 16 + n;
    const float bb = bv[c];
    bf16x8 o;
#pragma unroll
    for (int r = 0; r < 8; ++r) o[r] = (__bf16)(acc[t][r] + bb);
    *(bf16x8*)(vsT + ((size_t)(b * DKK + c)) * SS + s0 + half * 8) = o;
  }
}

// ---------------- K2: fused attention ----------------
// grid: B*(S/32) blocks, 256 threads (8 waves).
// LDS: scores [32][1024] f32 (128KB, reused for Wh at the end) + qs tile (4KB) + mean (8KB)
__launch_bounds__(256)
__global__ void attn_kernel(const __bf16* __restrict__ qs,
                            const __bf16* __restrict__ kkb,
                            const __bf16* __restrict__ vsT,
                            const float* __restrict__ Wh,
                            float* __restrict__ outp,
                            float* __restrict__ attn_out) {
  extern __shared__ char smem[];
  float*  sc    = (float*)smem;                                  // 32*1024 f32
  __bf16* qsl   = (__bf16*)(smem + 32 * 1024 * 4);               // 32*64 bf16
  float*  meanb = (float*)(smem + 32 * 1024 * 4 + 32 * 64 * 2);  // 32*64 f32

  const int tid = threadIdx.x, lane = tid & 31, wave = tid >> 5;
  const int b  = blockIdx.x >> 5;
  const int s0 = (blockIdx.x & 31) * 32;

  const int hm0 = (wave >> 2) * 16;
  const int hn0 = (wave & 3) * 16;
  v8f headAcc = {};

  const unsigned qslOff = (unsigned)(uintptr_t)qsl;  // LDS byte offset (low 32 bits)
  const unsigned scOff  = (unsigned)(uintptr_t)sc;

  for (int h = 0; h < HH; ++h) {
    // ---- stage qs row-panel into LDS via Tensor Data Mover (1-D, 4096B tile) ----
    if (wave == 0) {
      const __bf16* qsrc = qs + ((size_t)(b * HH + h) * SS + s0) * DKK;
      const unsigned long long ga = (unsigned long long)(uintptr_t)qsrc;
      // D# group0: count=1, lds_addr, global_addr[56:0], type=2
      u32x4 g0;
      g0.x = 1u;
      g0.y = qslOff;
      g0.z = (unsigned)ga;
      g0.w = (unsigned)((ga >> 32) & 0x01FFFFFFull) | (2u << 30);
      // D# group1: data_size=4B, tensor_dim0=1024, tensor_dim1=1,
      //            tile_dim0=1024 (1-D), tensor_dim0_stride=1024
      u32x8 g1;
      g1.s0 = 0x00020000u;      // workgroup_mask=0, data_size=2 (4B)
      g1.s1 = 1024u << 16;      // tensor_dim0[15:0]
      g1.s2 = 1u << 16;         // tensor_dim0[31:16]=0 | tensor_dim1[15:0]=1
      g1.s3 = 1024u << 16;      // tensor_dim1[31:16]=0 | tile_dim0=1024
      g1.s4 = 0u;               // tile_dim1=0 (unused), tile_dim2=0
      g1.s5 = 1024u;            // tensor_dim0_stride[31:0]
      g1.s6 = 0u;
      g1.s7 = 0u;
      asm volatile("tensor_load_to_lds %0, %1" :: "s"(g0), "s"(g1) : "memory");
      wait_tensorcnt0();
    }
    __syncthreads();

    // ---- scores = qs @ kk^T ----
    const __bf16* kbase = kkb + (size_t)(b * HH + h) * SS * DKK;
    for (int idx = wave; idx < 128; idx += 8) {
      const int m0 = (idx & 1) * 16;
      const int t0 = (idx >> 1) * 16;
      __builtin_prefetch(kbase + (size_t)(t0 + 128) * DKK, 0, 1);
      v8f acc = {};
#pragma unroll
      for (int kc = 0; kc < DKK; kc += 32) {
        v16bf a  = load_a_bf16(qsl, m0, DKK, kc, lane);
        v16bf bf = load_b_bf16(kbase, t0, DKK, kc, lane);
        acc = wmma_bf16(a, bf, acc);
      }
      const int n = lane & 15, half = lane >> 4;
#pragma unroll
      for (int r = 0; r < 8; ++r) {
        const int M = m0 + r + half * 8;
        const int t = t0 + n;
        const int sg = s0 + M;
        sc[M * 1024 + t] = (t > sg) ? NEG_INF : acc[r] * SCALE;
      }
    }
    __syncthreads();

    // ---- softmax per row (one wave per row); probs stay in LDS ----
    for (int m = wave; m < 32; m += 8) {
      float* row = sc + m * 1024;
      float mx = -3.0e38f;
      for (int t = lane; t < 1024; t += 32) mx = fmaxf(mx, row[t]);
#pragma unroll
      for (int off = 16; off; off >>= 1) mx = fmaxf(mx, __shfl_xor(mx, off, 32));
      float sum = 0.f;
      for (int t = lane; t < 1024; t += 32) {
        float e = __expf(row[t] - mx);
        row[t] = e;
        sum += e;
      }
#pragma unroll
      for (int off = 16; off; off >>= 1) sum += __shfl_xor(sum, off, 32);
      const float inv = 1.0f / sum;
      for (int t = lane; t < 1024; t += 32) row[t] *= inv;
    }
    __syncthreads();

    // ---- drain the 128KB attn panel LDS -> global on the ASYNC path ----
    // Panel is contiguous in LDS and contiguous in global (32 consecutive rows).
    {
      const char* gbase =
          (const char*)(attn_out + ((size_t)(b * HH + h) * SS + s0) * SS);
      for (int j = 0; j < 32; ++j) {
        const int i = wave * 32 + j;                 // 256 x 512B chunks
        const unsigned ldsA = scOff + (unsigned)(i * 512 + lane * 16);
        const char* gp = gbase + (size_t)i * 512 + lane * 16;
        asm volatile("global_store_async_from_lds_b128 %0, %1, off"
                     :: "v"(gp), "v"(ldsA) : "memory");
      }
    }

    // ---- head += attn @ vs (reads the same LDS panel; stores only read it too) ----
    const __bf16* vb = vsT + (size_t)b * DKK * SS;
    for (int tc = 0; tc < SS; tc += 32) {
      v16bf a  = load_a_f32(sc, hm0, 1024, tc, lane);
      v16bf bf = load_b_bf16(vb, hn0, SS, tc, lane);
      headAcc = wmma_bf16(a, bf, headAcc);
    }
    // async stores must complete before next head overwrites sc
    asm volatile("s_wait_asynccnt 0x0" ::: "memory");
    __syncthreads();
  }

  // mean over heads into LDS
  {
    const int n = lane & 15, half = lane >> 4;
#pragma unroll
    for (int r = 0; r < 8; ++r) {
      const int M = hm0 + r + half * 8;
      meanb[M * DKK + hn0 + n] = headAcc[r] * 0.25f;
    }
  }
  __syncthreads();

  // stage Wh (64KB) into the now-free scores LDS region
  {
    f32x4* dst = (f32x4*)sc;
    const f32x4* src = (const f32x4*)Wh;
    for (int i = tid; i < DKK * DD / 4; i += 256) dst[i] = src[i];
  }
  __syncthreads();

  // final out = mean @ Wh (fp32 VALU, operands from LDS)
  for (int e = tid; e < 32 * 256; e += 256) {
    const int m = e >> 8, d = e & 255;
    float acc = 0.f;
#pragma unroll 8
    for (int kk2 = 0; kk2 < DKK; ++kk2)
      acc += meanb[m * DKK + kk2] * sc[kk2 * 256 + d];
    outp[((size_t)b * SS + s0 + m) * 256 + d] = acc;
  }
}

// ---------------- host launcher ----------------
extern "C" void kernel_launch(void* const* d_in, const int* in_sizes, int n_in,
                              void* d_out, int out_size, void* d_ws, size_t ws_size,
                              hipStream_t stream) {
  const float* q  = (const float*)d_in[0];
  const float* k  = (const float*)d_in[1];
  const float* v  = (const float*)d_in[2];
  // d_in[3] = mask (causal, computed analytically)
  const float* Wq = (const float*)d_in[4];
  const float* bq = (const float*)d_in[5];
  const float* Wk = (const float*)d_in[6];
  const float* bk = (const float*)d_in[7];
  const float* Wv = (const float*)d_in[8];
  const float* bv = (const float*)d_in[9];
  const float* Wh = (const float*)d_in[10];

  float* outp = (float*)d_out;
  float* attn = outp + (size_t)BB * SS * DD;

  char* ws = (char*)d_ws;
  size_t off = 0;
  __bf16* qs_bf = (__bf16*)(ws + off); off += (size_t)BB * HH * SS * DKK * 2;
  __bf16* kk_bf = (__bf16*)(ws + off); off += (size_t)BB * HH * SS * DKK * 2;
  __bf16* vsT   = (__bf16*)(ws + off); off += (size_t)BB * DKK * SS * 2;
  __bf16* WqT   = (__bf16*)(ws + off); off += (size_t)HH * DKK * DD * 2;
  __bf16* WkT   = (__bf16*)(ws + off); off += (size_t)HH * DKK * DD * 2;
  __bf16* WvT   = (__bf16*)(ws + off); off += (size_t)DKK * DD * 2;

  {
    const int total = HH * DKK * DD + DKK * DD;  // 81920
    prep_weights_kernel<<<(total + 255) / 256, 256, 0, stream>>>(Wq, Wk, Wv,
                                                                 WqT, WkT, WvT);
  }
  proj_qk_kernel<<<dim3(BB * SS / 16, HH, 2), 64, 0, stream>>>(
      q, k, WqT, WkT, bq, bk, qs_bf, kk_bf);
  proj_v_kernel<<<BB * SS / 16, 64, 0, stream>>>(v, WvT, bv, vsT);

  const size_t lds = 32 * 1024 * 4 + 32 * DKK * 2 + 32 * DKK * 4;  // 143360 B
  attn_kernel<<<BB * (SS / 32), 256, lds, stream>>>(qs_bf, kk_bf, vsT, Wh,
                                                    outp, attn);
}
